// Block_24343874633736
// MI455X (gfx1250) — matrix operand order repaired and checked by reading, not compiled, not run on hardware
//
#include <hip/hip_runtime.h>

#define DIMD   1024
#define NHEAD  16
#define HDIM   64
#define NEXP   8
#define HIDD   1024
#define NTOK   4096   // B*S
#define SEQ    2048

typedef __attribute__((ext_vector_type(16))) __bf16 v16bf;
typedef __attribute__((ext_vector_type(8)))  __bf16 v8bf;
typedef __attribute__((ext_vector_type(4)))  __bf16 v4bf;
typedef __attribute__((ext_vector_type(8)))  float  v8f;

// ---- WMMA fragment loaders (layouts per CDNA5 ISA 7.12.2) ------------------
// A 16x32 bf16 (MxK): lanes 0-15 M=lane, K in [h*8,h*8+8) and [16+h*8,16+h*8+8)
__device__ __forceinline__ v16bf frag_a_p(const __bf16* rowp, int k0, int half) {
  union { v16bf v; v8bf h2[2]; } u;
  const __bf16* p = rowp + k0 + half * 8;
  u.h2[0] = *(const v8bf*)p;
  u.h2[1] = *(const v8bf*)(p + 16);
  return u.v;
}
// B 32x16 bf16 (KxN), stored N-major (Bt[N][K]): lane=N (mod 16), K in [h*8,h*8+8) and [16+h*8,16+h*8+8)
__device__ __forceinline__ v16bf frag_b_p(const __bf16* colp, int k0, int half) {
  union { v16bf v; v8bf h2[2]; } u;
  const __bf16* p = colp + k0 + half * 8;
  u.h2[0] = *(const v8bf*)p;
  u.h2[1] = *(const v8bf*)(p + 16);
  return u.v;
}
__device__ __forceinline__ v16bf frag_b(const __bf16* Bt, int ldb, int col, int k0, int half) {
  return frag_b_p(Bt + (size_t)col * ldb, k0, half);
}
#define WMMA_BF16(a, b, c) \
  __builtin_amdgcn_wmma_f32_16x16x32_bf16(false, (a), false, (b), (short)0, (c), false, false)

// ---- fp32 -> bf16 transposed weight prep -----------------------------------
__global__ void k_transpose_bf16(const float* __restrict__ in, __bf16* __restrict__ out,
                                 int R, int C, size_t in_stride, size_t out_stride) {
  in  += (size_t)blockIdx.z * in_stride;
  out += (size_t)blockIdx.z * out_stride;
  __shared__ float tile[32][33];
  int r0 = blockIdx.y * 32, c0 = blockIdx.x * 32;
  for (int i = threadIdx.y; i < 32; i += 8)
    tile[i][threadIdx.x] = in[(size_t)(r0 + i) * C + c0 + threadIdx.x];
  __syncthreads();
  for (int i = threadIdx.y; i < 32; i += 8)
    out[(size_t)(c0 + i) * R + r0 + threadIdx.x] = (__bf16)tile[threadIdx.x][i];
}

// ---- RMSNorm (fp32 in, bf16 out) -------------------------------------------
__global__ __launch_bounds__(256) void k_rmsnorm_bf16(const float* __restrict__ x,
                                                      const float* __restrict__ w,
                                                      __bf16* __restrict__ out) {
  int row = blockIdx.x;
  const float* xr = x + (size_t)row * DIMD;
  float s = 0.f;
  for (int d = threadIdx.x; d < DIMD; d += 256) { float v = xr[d]; s += v * v; }
#pragma unroll
  for (int off = 16; off; off >>= 1) s += __shfl_xor(s, off);
  __shared__ float wsum[8];
  if ((threadIdx.x & 31) == 0) wsum[threadIdx.x >> 5] = s;
  __syncthreads();
  float tot = 0.f;
#pragma unroll
  for (int i = 0; i < 8; ++i) tot += wsum[i];
  float r = rsqrtf(tot / (float)DIMD + 1e-6f);
  for (int d = threadIdx.x; d < DIMD; d += 256)
    out[(size_t)row * DIMD + d] = (__bf16)(xr[d] * r * w[d]);
}

// ---- WMMA GEMM: C(MxN,f32) = A(MxK,bf16) @ Bt(NxK,bf16)^T [+ resid] --------
// block = 128 thr (4 waves); wave computes 16 rows x 64 cols; grid (N/256, M/16)
// k-unrolled x2 with alternating register buffers: each fragment set is loaded
// and consumed in place (no ping-pong v_mov copies), loads for one half-step
// stay in flight behind the other half-step's WMMAs.
__global__ __launch_bounds__(128) void k_gemm_bf16(const __bf16* __restrict__ A,
                                                   const __bf16* __restrict__ Bt,
                                                   float* __restrict__ C,
                                                   const float* __restrict__ resid,
                                                   int M, int N, int K) {
  int wave = threadIdx.x >> 5, lane = threadIdx.x & 31;
  int lr = lane & 15, half = lane >> 4;
  int row0 = blockIdx.y * 16;
  int col0 = blockIdx.x * 256 + wave * 64;
  const __bf16* arow = A + (size_t)(row0 + lr) * K;
  const __bf16* bcol[4];
#pragma unroll
  for (int j = 0; j < 4; ++j) bcol[j] = Bt + (size_t)(col0 + j * 16 + lr) * K;

  v8f acc[4] = {};
  v16bf a0 = frag_a_p(arow, 0, half);
  v16bf b0[4];
#pragma unroll
  for (int j = 0; j < 4; ++j) b0[j] = frag_b_p(bcol[j], 0, half);

  int k0 = 0;
  for (; k0 < K - 64; k0 += 64) {
    // load buf1 @ k0+32, consume buf0 @ k0
    v16bf a1 = frag_a_p(arow, k0 + 32, half);
    v16bf b1[4];
#pragma unroll
    for (int j = 0; j < 4; ++j) b1[j] = frag_b_p(bcol[j], k0 + 32, half);
    __builtin_prefetch(bcol[0] + k0 + 192, 0, 1);   // pre-warm L2 ahead of pipeline
#pragma unroll
    for (int j = 0; j < 4; ++j) acc[j] = WMMA_BF16(a0, b0[j], acc[j]);
    // load buf0 @ k0+64, consume buf1 @ k0+32
    a0 = frag_a_p(arow, k0 + 64, half);
#pragma unroll
    for (int j = 0; j < 4; ++j) b0[j] = frag_b_p(bcol[j], k0 + 64, half);
#pragma unroll
    for (int j = 0; j < 4; ++j) acc[j] = WMMA_BF16(a1, b1[j], acc[j]);
  }
  // epilogue: k0 == K-64; buf0 holds K-64, load K-32 and finish
  {
    v16bf a1 = frag_a_p(arow, K - 32, half);
    v16bf b1[4];
#pragma unroll
    for (int j = 0; j < 4; ++j) b1[j] = frag_b_p(bcol[j], K - 32, half);
#pragma unroll
    for (int j = 0; j < 4; ++j) acc[j] = WMMA_BF16(a0, b0[j], acc[j]);
#pragma unroll
    for (int j = 0; j < 4; ++j) acc[j] = WMMA_BF16(a1, b1[j], acc[j]);
  }

#pragma unroll
  for (int j = 0; j < 4; ++j)
#pragma unroll
    for (int r = 0; r < 8; ++r) {
      int m = row0 + r + half * 8;
      int n = col0 + j * 16 + lr;
      float v = acc[j][r];
      if (resid) v += resid[(size_t)m * N + n];
      C[(size_t)m * N + n] = v;
    }
}

// ---- RoPE (in-place on fp32 q,k) -------------------------------------------
__global__ void k_rope(float* __restrict__ q, float* __restrict__ k,
                       const float* __restrict__ freqs) {
  size_t i = (size_t)blockIdx.x * blockDim.x + threadIdx.x;  // NTOK*NHEAD*32
  int p = (int)(i & 31);
  size_t t = i >> 5;
  int h = (int)(t & 15);
  size_t tok = t >> 4;
  int s = (int)(tok % SEQ);
  float c  = freqs[(size_t)s * 64 + p * 2];
  float sn = freqs[(size_t)s * 64 + p * 2 + 1];
  size_t base = tok * DIMD + h * HDIM + p * 2;
  float qr = q[base], qi = q[base + 1];
  q[base]     = qr * c - qi * sn;
  q[base + 1] = qr * sn + qi * c;
  float kr = k[base], ki = k[base + 1];
  k[base]     = kr * c - ki * sn;
  k[base + 1] = kr * sn + ki * c;
}

// ---- streaming-softmax attention: one wave per (token, head) ---------------
__global__ __launch_bounds__(256) void k_attention(const float* __restrict__ Q,
                                                   const float* __restrict__ Kk,
                                                   const float* __restrict__ V,
                                                   __bf16* __restrict__ O) {
  int wave = threadIdx.x >> 5, lane = threadIdx.x & 31;
  size_t gw = (size_t)blockIdx.x * 8 + wave;   // [0, NTOK*NHEAD)
  size_t tok = gw >> 4;
  int h = (int)(gw & 15);
  int s = (int)(tok % SEQ);
  const float* qp = Q + tok * DIMD + h * HDIM;
  float qv[HDIM];
#pragma unroll
  for (int d = 0; d < HDIM; ++d) qv[d] = qp[d] * 0.125f;   // 1/sqrt(64)
  float m = -3.0e38f, l = 0.f;
  float acc[HDIM];
#pragma unroll
  for (int d = 0; d < HDIM; ++d) acc[d] = 0.f;
  size_t base = (tok - (size_t)s) * DIMD + h * HDIM;       // batch start
  for (int kk = lane; kk < SEQ; kk += 32) {
    const float4* kp = (const float4*)(Kk + base + (size_t)kk * DIMD);
    float sc = 0.f;
#pragma unroll
    for (int d4 = 0; d4 < HDIM / 4; ++d4) {
      float4 kv = kp[d4];
      sc += qv[4 * d4]     * kv.x + qv[4 * d4 + 1] * kv.y +
            qv[4 * d4 + 2] * kv.z + qv[4 * d4 + 3] * kv.w;
    }
    float mn = fmaxf(m, sc);
    float corr = __expf(m - mn);
    float pw = __expf(sc - mn);
    const float4* vp = (const float4*)(V + base + (size_t)kk * DIMD);
#pragma unroll
    for (int d4 = 0; d4 < HDIM / 4; ++d4) {
      float4 vv = vp[d4];
      acc[4 * d4]     = acc[4 * d4]     * corr + pw * vv.x;
      acc[4 * d4 + 1] = acc[4 * d4 + 1] * corr + pw * vv.y;
      acc[4 * d4 + 2] = acc[4 * d4 + 2] * corr + pw * vv.z;
      acc[4 * d4 + 3] = acc[4 * d4 + 3] * corr + pw * vv.w;
    }
    l = l * corr + pw;
    m = mn;
  }
#pragma unroll
  for (int off = 16; off; off >>= 1) {
    float mo = __shfl_xor(m, off);
    float lo = __shfl_xor(l, off);
    float mn = fmaxf(m, mo);
    float c1 = __expf(m - mn), c2 = __expf(mo - mn);
#pragma unroll
    for (int d = 0; d < HDIM; ++d)
      acc[d] = acc[d] * c1 + __shfl_xor(acc[d], off) * c2;
    l = l * c1 + lo * c2;
    m = mn;
  }
  float inv = 1.f / l;
  if (lane == 0) {
    __bf16* op = O + tok * DIMD + h * HDIM;
#pragma unroll
    for (int d = 0; d < HDIM; ++d) op[d] = (__bf16)(acc[d] * inv);
  }
}

// ---- gating: softmax over 8 experts, top-2, build per-expert token lists ---
__global__ void k_zero_i32(int* p, int n) {
  int i = blockIdx.x * blockDim.x + threadIdx.x;
  if (i < n) p[i] = 0;
}

__global__ void k_gate(const __bf16* __restrict__ hn, const float* __restrict__ gw,
                       float* __restrict__ tw, int* __restrict__ tidx,
                       int* __restrict__ elist, int* __restrict__ ejlist,
                       int* __restrict__ counts) {
  int n = blockIdx.x * blockDim.x + threadIdx.x;
  if (n >= NTOK) return;
  const __bf16* hp = hn + (size_t)n * DIMD;
  float s[NEXP];
#pragma unroll
  for (int e = 0; e < NEXP; ++e) s[e] = 0.f;
  for (int d = 0; d < DIMD; ++d) {
    float hv = (float)hp[d];
#pragma unroll
    for (int e = 0; e < NEXP; ++e) s[e] += hv * gw[e * DIMD + d];
  }
  float mx = s[0];
#pragma unroll
  for (int e = 1; e < NEXP; ++e) mx = fmaxf(mx, s[e]);
  float den = 0.f;
#pragma unroll
  for (int e = 0; e < NEXP; ++e) { s[e] = __expf(s[e] - mx); den += s[e]; }
  float dinv = 1.f / den;
#pragma unroll
  for (int e = 0; e < NEXP; ++e) s[e] *= dinv;
  int i0 = 0;
#pragma unroll
  for (int e = 1; e < NEXP; ++e) if (s[e] > s[i0]) i0 = e;
  int i1 = (i0 == 0) ? 1 : 0;
#pragma unroll
  for (int e = 0; e < NEXP; ++e) if (e != i0 && s[e] > s[i1]) i1 = e;
  tidx[n * 2 + 0] = i0; tw[n * 2 + 0] = s[i0];
  tidx[n * 2 + 1] = i1; tw[n * 2 + 1] = s[i1];
  int p0 = atomicAdd(&counts[i0], 1);
  elist[(size_t)i0 * NTOK + p0] = n; ejlist[(size_t)i0 * NTOK + p0] = 0;
  int p1 = atomicAdd(&counts[i1], 1);
  elist[(size_t)i1 * NTOK + p1] = n; ejlist[(size_t)i1 * NTOK + p1] = 1;
}

// ---- swiglu elementwise (shared expert), b128 I/O --------------------------
__global__ void k_swiglu(const float* __restrict__ h1, const float* __restrict__ h3,
                         __bf16* __restrict__ out, size_t total4) {
  size_t i = (size_t)blockIdx.x * blockDim.x + threadIdx.x;
  if (i >= total4) return;
  float4 a = ((const float4*)h1)[i];
  float4 b = ((const float4*)h3)[i];
  v4bf o;
  o[0] = (__bf16)((a.x / (1.f + __expf(-a.x))) * b.x);
  o[1] = (__bf16)((a.y / (1.f + __expf(-a.y))) * b.y);
  o[2] = (__bf16)((a.z / (1.f + __expf(-a.z))) * b.z);
  o[3] = (__bf16)((a.w / (1.f + __expf(-a.w))) * b.w);
  *(v4bf*)(out + i * 4) = o;
}

// ---- fused routed-expert FFN: per (expert, 16-token tile) workgroup --------
// phase1: act(16,HIDD) = silu(hn_g @ w1) * (hn_g @ w3) -> LDS (bf16)
// phase2: eo_slot = (act @ w2) * route_weight, scattered by (token, topk-slot)
// Loads are batched 8-at-a-time per k-step so WMMAs issue back-to-back.
__global__ __launch_bounds__(256) void k_expert_ffn(
    const __bf16* __restrict__ hn, const __bf16* __restrict__ ew1T,
    const __bf16* __restrict__ ew3T, const __bf16* __restrict__ ew2T,
    const float* __restrict__ tw, const int* __restrict__ elist,
    const int* __restrict__ ejlist, const int* __restrict__ counts,
    float* __restrict__ eo) {
  int e = blockIdx.y;
  int cnt = counts[e];
  int r0 = blockIdx.x * 16;
  if (r0 >= cnt) return;
  __shared__ __bf16 act[16][HIDD + 8];
  __shared__ int tk[16];
  __shared__ int tj[16];
  __shared__ float wr[16];
  if (threadIdx.x < 16) {
    int p = r0 + (int)threadIdx.x;
    if (p < cnt) {
      int t = elist[(size_t)e * NTOK + p];
      int j = ejlist[(size_t)e * NTOK + p];
      tk[threadIdx.x] = t;
      tj[threadIdx.x] = j;
      wr[threadIdx.x] = tw[(size_t)t * 2 + j];
    } else {
      tk[threadIdx.x] = elist[(size_t)e * NTOK + r0];  // valid row for safe loads
      tj[threadIdx.x] = -1;
      wr[threadIdx.x] = 0.f;
    }
  }
  __syncthreads();
  int wave = threadIdx.x >> 5, lane = threadIdx.x & 31;
  int lr = lane & 15, half = lane >> 4;
  int colbase = wave * 128;
  const __bf16* w1 = ew1T + (size_t)e * HIDD * DIMD;
  const __bf16* w3 = ew3T + (size_t)e * HIDD * DIMD;
  const __bf16* w2 = ew2T + (size_t)e * DIMD * HIDD;
  const __bf16* hrow = hn + (size_t)tk[lr] * DIMD;  // gathered token row
  v8f a1[8] = {};
  v8f a3[8] = {};
  for (int k0 = 0; k0 < DIMD; k0 += 32) {
    v16bf af = frag_a_p(hrow, k0, half);
#pragma unroll
    for (int g = 0; g < 2; ++g) {
      v16bf bb1[4], bb3[4];
#pragma unroll
      for (int j = 0; j < 4; ++j) {
        int col = colbase + (g * 4 + j) * 16 + lr;
        bb1[j] = frag_b(w1, DIMD, col, k0, half);
        bb3[j] = frag_b(w3, DIMD, col, k0, half);
      }
#pragma unroll
      for (int j = 0; j < 4; ++j) {
        a1[g * 4 + j] = WMMA_BF16(af, bb1[j], a1[g * 4 + j]);
        a3[g * 4 + j] = WMMA_BF16(af, bb3[j], a3[g * 4 + j]);
      }
    }
  }
#pragma unroll
  for (int j = 0; j < 8; ++j)
#pragma unroll
    for (int r = 0; r < 8; ++r) {
      int row = r + half * 8;
      int col = colbase + j * 16 + lr;
      float g = a1[j][r];
      act[row][col] = (__bf16)((g / (1.f + __expf(-g))) * a3[j][r]);
    }
  __syncthreads();
  v8f o[8] = {};
  for (int k0 = 0; k0 < HIDD; k0 += 32) {
    union { v16bf v; v8bf h2[2]; } ua;
    const __bf16* ap = &act[lr][k0 + half * 8];
    ua.h2[0] = *(const v8bf*)ap;
    ua.h2[1] = *(const v8bf*)(ap + 16);
#pragma unroll
    for (int g = 0; g < 2; ++g) {
      v16bf bb[4];
#pragma unroll
      for (int j = 0; j < 4; ++j)
        bb[j] = frag_b(w2, HIDD, colbase + (g * 4 + j) * 16 + lr, k0, half);
#pragma unroll
      for (int j = 0; j < 4; ++j)
        o[g * 4 + j] = WMMA_BF16(ua.v, bb[j], o[g * 4 + j]);
    }
  }
#pragma unroll
  for (int j = 0; j < 8; ++j)
#pragma unroll
    for (int r = 0; r < 8; ++r) {
      int row = r + half * 8;
      if (tj[row] >= 0) {
        size_t off = ((size_t)tk[row] * 2 + tj[row]) * DIMD + colbase + j * 16 + lr;
        eo[off] = o[j][r] * wr[row];
      }
    }
}

// ---- final combine: out = h + shared + eo[slot0] + eo[slot1] (b128 I/O) ----
__global__ void k_combine(const float* __restrict__ h, const float* __restrict__ sh,
                          const float* __restrict__ eo, float* __restrict__ out) {
  size_t i4 = (size_t)blockIdx.x * blockDim.x + threadIdx.x;  // NTOK*DIMD/4
  size_t i = i4 * 4;
  size_t n = i >> 10;
  int d = (int)(i & 1023);
  float4 hv = ((const float4*)h)[i4];
  float4 sv = ((const float4*)sh)[i4];
  float4 e0 = *(const float4*)(eo + (n * 2) * DIMD + d);
  float4 e1 = *(const float4*)(eo + (n * 2 + 1) * DIMD + d);
  float4 r;
  r.x = hv.x + sv.x + e0.x + e1.x;
  r.y = hv.y + sv.y + e0.y + e1.y;
  r.z = hv.z + sv.z + e0.z + e1.z;
  r.w = hv.w + sv.w + e0.w + e1.w;
  ((float4*)out)[i4] = r;
}

// ---------------------------------------------------------------------------
extern "C" void kernel_launch(void* const* d_in, const int* in_sizes, int n_in,
                              void* d_out, int out_size, void* d_ws, size_t ws_size,
                              hipStream_t stream) {
  (void)in_sizes; (void)n_in; (void)out_size; (void)ws_size;
  const float* x     = (const float*)d_in[0];
  const float* freqs = (const float*)d_in[1];
  const float* anw   = (const float*)d_in[2];
  const float* wq    = (const float*)d_in[3];
  const float* wk    = (const float*)d_in[4];
  const float* wv    = (const float*)d_in[5];
  const float* wo    = (const float*)d_in[6];
  const float* fnw   = (const float*)d_in[7];
  const float* gatew = (const float*)d_in[8];
  const float* ew1   = (const float*)d_in[9];
  const float* ew2   = (const float*)d_in[10];
  const float* ew3   = (const float*)d_in[11];
  const float* sw1   = (const float*)d_in[12];
  const float* sw2   = (const float*)d_in[13];
  const float* sw3   = (const float*)d_in[14];
  float* out = (float*)d_out;

  char* ws = (char*)d_ws;
  size_t off = 0;
  auto alloc = [&](size_t bytes) -> char* {
    char* p = ws + off;
    off += (bytes + 255) & ~(size_t)255;
    return p;
  };
  const size_t MM = (size_t)1024 * 1024;
  __bf16* wqT   = (__bf16*)alloc(MM * 2);
  __bf16* wkT   = (__bf16*)alloc(MM * 2);
  __bf16* wvT   = (__bf16*)alloc(MM * 2);
  __bf16* woT   = (__bf16*)alloc(MM * 2);
  __bf16* sw1T  = (__bf16*)alloc(MM * 2);
  __bf16* sw3T  = (__bf16*)alloc(MM * 2);
  __bf16* sw2T  = (__bf16*)alloc(MM * 2);
  __bf16* ew1T  = (__bf16*)alloc(8 * MM * 2);
  __bf16* ew3T  = (__bf16*)alloc(8 * MM * 2);
  __bf16* ew2T  = (__bf16*)alloc(8 * MM * 2);
  __bf16* xn    = (__bf16*)alloc((size_t)NTOK * DIMD * 2);
  float*  qf    = (float*)alloc((size_t)NTOK * DIMD * 4);
  float*  kf    = (float*)alloc((size_t)NTOK * DIMD * 4);
  float*  vf    = (float*)alloc((size_t)NTOK * DIMD * 4);
  __bf16* attb  = (__bf16*)alloc((size_t)NTOK * DIMD * 2);
  float*  hbuf  = (float*)alloc((size_t)NTOK * DIMD * 4);
  __bf16* hnb   = (__bf16*)alloc((size_t)NTOK * DIMD * 2);
  float*  h1s   = (float*)alloc((size_t)NTOK * HIDD * 4);
  float*  h3s   = (float*)alloc((size_t)NTOK * HIDD * 4);
  __bf16* s1b   = (__bf16*)alloc((size_t)NTOK * HIDD * 2);
  float*  shbuf = (float*)alloc((size_t)NTOK * DIMD * 4);
  float*  eo    = (float*)alloc((size_t)NTOK * 2 * DIMD * 4);
  float*  tww   = (float*)alloc((size_t)NTOK * 2 * 4);
  int*    tidx  = (int*)alloc((size_t)NTOK * 2 * 4);
  int*    elist = (int*)alloc((size_t)NEXP * NTOK * 4);
  int*    ejlst = (int*)alloc((size_t)NEXP * NTOK * 4);
  int*    cnts  = (int*)alloc(256);

  dim3 tb(32, 8);
  dim3 tg(32, 32, 1);
  // weight prep: bf16 + N-major (so WMMA B-fragments are contiguous per lane)
  k_transpose_bf16<<<tg, tb, 0, stream>>>(wq,  wqT,  1024, 1024, 0, 0);
  k_transpose_bf16<<<tg, tb, 0, stream>>>(wk,  wkT,  1024, 1024, 0, 0);
  k_transpose_bf16<<<tg, tb, 0, stream>>>(wv,  wvT,  1024, 1024, 0, 0);
  k_transpose_bf16<<<tg, tb, 0, stream>>>(wo,  woT,  1024, 1024, 0, 0);
  k_transpose_bf16<<<tg, tb, 0, stream>>>(sw1, sw1T, 1024, 1024, 0, 0);
  k_transpose_bf16<<<tg, tb, 0, stream>>>(sw3, sw3T, 1024, 1024, 0, 0);
  k_transpose_bf16<<<tg, tb, 0, stream>>>(sw2, sw2T, 1024, 1024, 0, 0);
  dim3 tge(32, 32, 8);
  k_transpose_bf16<<<tge, tb, 0, stream>>>(ew1, ew1T, 1024, 1024, MM, MM);
  k_transpose_bf16<<<tge, tb, 0, stream>>>(ew3, ew3T, 1024, 1024, MM, MM);
  k_transpose_bf16<<<tge, tb, 0, stream>>>(ew2, ew2T, 1024, 1024, MM, MM);

  // attention sub-block
  k_rmsnorm_bf16<<<NTOK, 256, 0, stream>>>(x, anw, xn);
  dim3 gg(DIMD / 256, NTOK / 16);
  k_gemm_bf16<<<gg, 128, 0, stream>>>(xn, wqT, qf, nullptr, NTOK, DIMD, DIMD);
  k_gemm_bf16<<<gg, 128, 0, stream>>>(xn, wkT, kf, nullptr, NTOK, DIMD, DIMD);
  k_gemm_bf16<<<gg, 128, 0, stream>>>(xn, wvT, vf, nullptr, NTOK, DIMD, DIMD);
  k_rope<<<(NTOK * NHEAD * 32) / 256, 256, 0, stream>>>(qf, kf, freqs);
  k_attention<<<(NTOK * NHEAD) / 8, 256, 0, stream>>>(qf, kf, vf, attb);
  k_gemm_bf16<<<gg, 128, 0, stream>>>(attb, woT, hbuf, x, NTOK, DIMD, DIMD);  // h = x + att@wo

  // MoE sub-block
  k_rmsnorm_bf16<<<NTOK, 256, 0, stream>>>(hbuf, fnw, hnb);
  k_zero_i32<<<1, 32, 0, stream>>>(cnts, NEXP);
  k_gate<<<NTOK / 256, 256, 0, stream>>>(hnb, gatew, tww, tidx, elist, ejlst, cnts);
  // shared expert
  k_gemm_bf16<<<gg, 128, 0, stream>>>(hnb, sw1T, h1s, nullptr, NTOK, HIDD, DIMD);
  k_gemm_bf16<<<gg, 128, 0, stream>>>(hnb, sw3T, h3s, nullptr, NTOK, HIDD, DIMD);
  k_swiglu<<<(NTOK * HIDD / 4) / 256, 256, 0, stream>>>(h1s, h3s, s1b, (size_t)NTOK * HIDD / 4);
  k_gemm_bf16<<<gg, 128, 0, stream>>>(s1b, sw2T, shbuf, nullptr, NTOK, DIMD, HIDD);
  // routed experts (top-2 only; blocks past count[e] early-exit)
  k_expert_ffn<<<dim3(NTOK / 16, NEXP), 256, 0, stream>>>(hnb, ew1T, ew3T, ew2T,
                                                          tww, elist, ejlst, cnts, eo);
  // out = h + shared + routed
  k_combine<<<(NTOK * DIMD / 4) / 256, 256, 0, stream>>>(hbuf, shbuf, eo, out);
}
